// SinkhornRouter_24215025615341
// MI455X (gfx1250) — matrix operand order repaired and checked by reading, not compiled, not run on hardware
//
#include <hip/hip_runtime.h>
#include <hip/hip_bf16.h>
#include <stdint.h>

typedef __attribute__((ext_vector_type(16))) _Float16 v16h;
typedef __attribute__((ext_vector_type(8)))  _Float16 v8h;
typedef __attribute__((ext_vector_type(8)))  float    v8f;

#define NB   4
#define NS   2048
#define NH   2048
#define NE   64
#define NM   8192          // B*S
#define CAPF 160.0f        // 1.25 * 8192 / 64
#define EPSF 1e-9f

// ---------------------------------------------------------------------------
// CDNA5 async global->LDS copy (ASYNCcnt-tracked, bypasses VGPRs).
// lds: wave-relative LDS byte address; gaddr: 64-bit global byte address.
// ---------------------------------------------------------------------------
__device__ inline void async_ld16(uint32_t lds, uint64_t gaddr) {
  asm volatile("global_load_async_to_lds_b128 %0, %1, off"
               :: "v"(lds), "v"(gaddr) : "memory");
}

#if __has_builtin(__builtin_amdgcn_s_wait_asynccnt)
#define WAIT_ASYNC(n) __builtin_amdgcn_s_wait_asynccnt(n)
#else
#define WAIT_ASYNC(n) asm volatile("s_wait_asynccnt %0" :: "n"(n) : "memory")
#endif

__device__ inline uint32_t lds_addr(const void* p) {
  return (uint32_t)(uintptr_t)p;   // low 32 bits of generic LDS ptr = LDS offset
}

// ---------------------------------------------------------------------------
// Transpose f32 [R][C] -> f16 [C][R], 64x64 LDS tiles. R,C multiples of 64.
// ---------------------------------------------------------------------------
__global__ __launch_bounds__(256) void k_transpose_f32_f16(
    const float* __restrict__ in, _Float16* __restrict__ out, int R, int C) {
  __shared__ float t[64][65];
  const int cb = blockIdx.x * 64, rb = blockIdx.y * 64;
  const int tid = threadIdx.x;
#pragma unroll
  for (int j = 0; j < 16; ++j) {
    int idx = tid + j * 256;
    int r = idx >> 6, c = idx & 63;
    t[r][c] = in[(size_t)(rb + r) * C + cb + c];
  }
  __syncthreads();
#pragma unroll
  for (int j = 0; j < 16; ++j) {
    int idx = tid + j * 256;
    int r = idx >> 6, c = idx & 63;
    out[(size_t)(cb + r) * R + rb + c] = (_Float16)t[c][r];
  }
}

// ---------------------------------------------------------------------------
// LayerNorm over H=2048, one block (256 thr) per row, f16 output.
// ---------------------------------------------------------------------------
__global__ __launch_bounds__(256) void k_layernorm(
    const float* __restrict__ x, const float* __restrict__ gamma,
    const float* __restrict__ beta, _Float16* __restrict__ xn) {
  __shared__ float r1[256], r2[256];
  const int tid = threadIdx.x;
  const size_t base = (size_t)blockIdx.x * NH;
  float v[8], s = 0.f, sq = 0.f;
#pragma unroll
  for (int j = 0; j < 8; ++j) {
    v[j] = x[base + tid + j * 256];
    s += v[j]; sq += v[j] * v[j];
  }
  r1[tid] = s; r2[tid] = sq;
  __syncthreads();
  for (int off = 128; off > 0; off >>= 1) {
    if (tid < off) { r1[tid] += r1[tid + off]; r2[tid] += r2[tid + off]; }
    __syncthreads();
  }
  const float mean = r1[0] * (1.0f / NH);
  const float var  = fmaxf(r2[0] * (1.0f / NH) - mean * mean, 0.0f);
  const float rstd = rsqrtf(var + 1e-5f);
#pragma unroll
  for (int j = 0; j < 8; ++j) {
    int i = tid + j * 256;
    float y = (v[j] - mean) * rstd * gamma[i] + beta[i];
    xn[base + i] = (_Float16)y;
  }
}

// ---------------------------------------------------------------------------
// WMMA fragment loaders from padded LDS (row stride 40 halves = 80B).
// A 16x32 f16: lane<16 -> K{0..7,16..23}; lane>=16 -> K{8..15,24..31}.
// B 32x16 f16 (from Bt, col-major): same K split as A.
// ---------------------------------------------------------------------------
__device__ inline v16h ld_frag_a(const _Float16* As, int row, int hs) {
  const _Float16* p = As + row * 40 + hs * 8;
  v8h lo = *(const v8h*)p;
  v8h hi = *(const v8h*)(p + 16);
  v16h r;
#pragma unroll
  for (int i = 0; i < 8; ++i) { r[i] = lo[i]; r[i + 8] = hi[i]; }
  return r;
}
__device__ inline v16h ld_frag_b(const _Float16* Bs, int row, int hs) {
  const _Float16* p = Bs + row * 40 + hs * 8;
  v8h lo = *(const v8h*)p;
  v8h hi = *(const v8h*)(p + 16);
  v16h r;
#pragma unroll
  for (int i = 0; i < 8; ++i) { r[i] = lo[i]; r[i + 8] = hi[i]; }
  return r;
}

// ---------------------------------------------------------------------------
// WMMA GEMM: C[M][N] = act( A[M][K](f16) * Bt[N][K](f16)^T + bias ) * scale
// Block tile 128x64, 256 threads = 8 waves, wave tile 32x32 (2x2 WMMA).
// Double-buffered LDS, async global->LDS staging, last iteration peeled so
// the steady-state loop is branch-free.
// ---------------------------------------------------------------------------
template <bool RELU, bool OUT16>
__global__ __launch_bounds__(256) void k_gemm_wmma(
    const _Float16* __restrict__ A, const _Float16* __restrict__ Bt,
    const float* __restrict__ bias, const float* __restrict__ tau,
    void* __restrict__ outp, int M, int N, int K) {
  __shared__ _Float16 As[2][128 * 40];
  __shared__ _Float16 Bs[2][64 * 40];
  const int tid  = threadIdx.x;
  const int lane = tid & 31, wid = tid >> 5;
  const int wm0 = (wid & 3) * 32, wn0 = (wid >> 2) * 32;
  const int gm0 = blockIdx.y * 128, gn0 = blockIdx.x * 64;
  const int hs = lane >> 4, lr = lane & 15;

  // Per-thread staging slots: A gets 2 x 8-half chunks, B gets 1.
  const int cA0 = tid, cA1 = tid + 256;
  const int rA0 = cA0 >> 2, oA0 = (cA0 & 3) * 8;
  const int rA1 = cA1 >> 2, oA1 = (cA1 & 3) * 8;
  const int rB  = tid >> 2, oB  = (tid & 3) * 8;

  const uint64_t gA0 = (uint64_t)(uintptr_t)&A[(size_t)(gm0 + rA0) * K + oA0];
  const uint64_t gA1 = (uint64_t)(uintptr_t)&A[(size_t)(gm0 + rA1) * K + oA1];
  const uint64_t gB  = (uint64_t)(uintptr_t)&Bt[(size_t)(gn0 + rB) * K + oB];

  uint32_t lA0[2], lA1[2], lB[2];
#pragma unroll
  for (int b = 0; b < 2; ++b) {
    lA0[b] = lds_addr(&As[b][rA0 * 40 + oA0]);
    lA1[b] = lds_addr(&As[b][rA1 * 40 + oA1]);
    lB[b]  = lds_addr(&Bs[b][rB * 40 + oB]);
  }

  v8f c[2][2] = {};

  auto compute = [&](int buf) {
    const _Float16* Ab = As[buf];
    const _Float16* Bb = Bs[buf];
    v16h a0 = ld_frag_a(Ab, wm0 + lr,      hs);
    v16h a1 = ld_frag_a(Ab, wm0 + 16 + lr, hs);
    v16h b0 = ld_frag_b(Bb, wn0 + lr,      hs);
    v16h b1 = ld_frag_b(Bb, wn0 + 16 + lr, hs);
    c[0][0] = __builtin_amdgcn_wmma_f32_16x16x32_f16(false, a0, false, b0, (short)0, c[0][0], false, false);
    c[0][1] = __builtin_amdgcn_wmma_f32_16x16x32_f16(false, a0, false, b1, (short)0, c[0][1], false, false);
    c[1][0] = __builtin_amdgcn_wmma_f32_16x16x32_f16(false, a1, false, b0, (short)0, c[1][0], false, false);
    c[1][1] = __builtin_amdgcn_wmma_f32_16x16x32_f16(false, a1, false, b1, (short)0, c[1][1], false, false);
  };

  const int T = K / 32;
  // Prologue: stage tile 0 into buffer 0 (k-step stride = 32 halves = 64B).
  async_ld16(lA0[0], gA0);
  async_ld16(lA1[0], gA1);
  async_ld16(lB[0],  gB);

  // Steady state: branch-free (last iteration peeled).
  for (int t = 0; t < T - 1; ++t) {
    const int cur = t & 1;
    const uint64_t off = (uint64_t)(t + 1) * 64;
    async_ld16(lA0[cur ^ 1], gA0 + off);
    async_ld16(lA1[cur ^ 1], gA1 + off);
    async_ld16(lB[cur ^ 1],  gB + off);
    WAIT_ASYNC(3);               // current tile's 3 ops complete
    __syncthreads();             // all waves' staging visible
    compute(cur);
    __syncthreads();             // done reading 'cur' before next overwrite
  }
  // Epilogue: last tile.
  WAIT_ASYNC(0);
  __syncthreads();
  compute((T - 1) & 1);

  float sc = 1.0f;
  if (tau) sc = 1.0f / fmaxf(*tau, 1e-6f);

#pragma unroll
  for (int i = 0; i < 2; ++i)
#pragma unroll
    for (int j = 0; j < 2; ++j) {
      const int n = gn0 + wn0 + j * 16 + lr;
      const float bv = bias[n];
#pragma unroll
      for (int v = 0; v < 8; ++v) {
        const int m = gm0 + wm0 + i * 16 + v + 8 * hs;
        float val = c[i][j][v] + bv;
        if (RELU) val = fmaxf(val, 0.0f);
        val *= sc;
        if (OUT16) ((_Float16*)outp)[(size_t)m * N + n] = (_Float16)val;
        else       ((float*)outp)[(size_t)m * N + n] = val;
      }
    }
}

// ---------------------------------------------------------------------------
// scores = exp(logits - rowmax). One wave per row (E=64, 2 elems/lane).
// ---------------------------------------------------------------------------
__global__ __launch_bounds__(256) void k_scores(float* __restrict__ a) {
  const int lane = threadIdx.x & 31;
  const int row  = blockIdx.x * 8 + (threadIdx.x >> 5);
  const size_t base = (size_t)row * NE;
  float v0 = a[base + lane], v1 = a[base + lane + 32];
  float m = fmaxf(v0, v1);
#pragma unroll
  for (int off = 16; off > 0; off >>= 1) m = fmaxf(m, __shfl_xor(m, off, 32));
  a[base + lane]      = __expf(v0 - m);
  a[base + lane + 32] = __expf(v1 - m);
}

// ---------------------------------------------------------------------------
// Row normalize: a /= (rowsum + eps). One wave per row.
// ---------------------------------------------------------------------------
__global__ __launch_bounds__(256) void k_rownorm(float* __restrict__ a) {
  const int lane = threadIdx.x & 31;
  const int row  = blockIdx.x * 8 + (threadIdx.x >> 5);
  const size_t base = (size_t)row * NE;
  float v0 = a[base + lane], v1 = a[base + lane + 32];
  float s = v0 + v1;
#pragma unroll
  for (int off = 16; off > 0; off >>= 1) s += __shfl_xor(s, off, 32);
  const float inv = 1.0f / (s + EPSF);
  a[base + lane]      = v0 * inv;
  a[base + lane + 32] = v1 * inv;
}

// ---------------------------------------------------------------------------
// Column op over axis S: one block per (b,e). STATS: write (colsum-cap)^2.
// Otherwise scale column by cap/(colsum+eps). Fixed-order reduction.
// ---------------------------------------------------------------------------
template <bool STATS>
__global__ __launch_bounds__(256) void k_col(float* __restrict__ a,
                                             float* __restrict__ colaux) {
  __shared__ float red[256];
  const int tid = threadIdx.x;
  const int b = blockIdx.x >> 6, e = blockIdx.x & 63;
  const size_t base = ((size_t)b * NS) * NE + e;
  float s = 0.f;
#pragma unroll
  for (int j = 0; j < 8; ++j) s += a[base + (size_t)(tid + j * 256) * NE];
  red[tid] = s;
  __syncthreads();
  for (int off = 128; off > 0; off >>= 1) {
    if (tid < off) red[tid] += red[tid + off];
    __syncthreads();
  }
  const float cs = red[0];
  if (STATS) {
    if (tid == 0) colaux[blockIdx.x] = (cs - CAPF) * (cs - CAPF);
  } else {
    const float sc = CAPF / (cs + EPSF);
#pragma unroll
    for (int j = 0; j < 8; ++j) a[base + (size_t)(tid + j * 256) * NE] *= sc;
  }
}

// ---------------------------------------------------------------------------
// Final: rowsum/aux, router_probs, top-2 (tie -> lower index), renorm, write.
// One wave per row. Indices written as float(idx) into the float output.
// ---------------------------------------------------------------------------
__global__ __launch_bounds__(256) void k_final(const float* __restrict__ a,
                                               float* __restrict__ rowaux,
                                               float* __restrict__ oidx,
                                               float* __restrict__ oprob) {
  const int lane = threadIdx.x & 31;
  const int row  = blockIdx.x * 8 + (threadIdx.x >> 5);
  const size_t base = (size_t)row * NE;
  float a0 = a[base + lane], a1 = a[base + lane + 32];
  float rs = a0 + a1;
#pragma unroll
  for (int off = 16; off > 0; off >>= 1) rs += __shfl_xor(rs, off, 32);
  if (lane == 0) rowaux[row] = (rs - 1.0f) * (rs - 1.0f);
  const float inv = 1.0f / (rs + EPSF);
  const float p0 = a0 * inv, p1 = a1 * inv;

  float v1, v2; int i1, i2;
  if (p1 > p0) { v1 = p1; i1 = lane + 32; v2 = p0; i2 = lane; }
  else         { v1 = p0; i1 = lane;      v2 = p1; i2 = lane + 32; }
#pragma unroll
  for (int off = 16; off > 0; off >>= 1) {
    float ov1 = __shfl_xor(v1, off, 32), ov2 = __shfl_xor(v2, off, 32);
    int   oi1 = __shfl_xor(i1, off, 32), oi2 = __shfl_xor(i2, off, 32);
    bool ofirst = (ov1 > v1) || (ov1 == v1 && oi1 < i1);
    float nv1 = ofirst ? ov1 : v1;  int ni1 = ofirst ? oi1 : i1;
    float cv  = ofirst ? v1  : ov1; int ci  = ofirst ? i1  : oi1;  // loser of top1
    float wv  = ofirst ? ov2 : v2;  int wi  = ofirst ? oi2 : i2;   // winner's 2nd
    bool cwin = (cv > wv) || (cv == wv && ci < wi);
    v1 = nv1; i1 = ni1;
    v2 = cwin ? cv : wv; i2 = cwin ? ci : wi;
  }
  if (lane == 0) {
    const float dn = 1.0f / (v1 + v2 + EPSF);
    oidx[(size_t)row * 2]     = (float)i1;
    oidx[(size_t)row * 2 + 1] = (float)i2;
    oprob[(size_t)row * 2]     = v1 * dn;
    oprob[(size_t)row * 2 + 1] = v2 * dn;
  }
}

// ---------------------------------------------------------------------------
// aux = mean(rowaux[8192]) + mean(colaux[256]); single block, fixed order.
// ---------------------------------------------------------------------------
__global__ __launch_bounds__(256) void k_aux(const float* __restrict__ rowaux,
                                             const float* __restrict__ colaux,
                                             float* __restrict__ out) {
  __shared__ float red[256];
  const int tid = threadIdx.x;
  float s = 0.f;
  for (int j = 0; j < 32; ++j) s += rowaux[tid + j * 256];
  red[tid] = s;
  __syncthreads();
  for (int off = 128; off > 0; off >>= 1) {
    if (tid < off) red[tid] += red[tid + off];
    __syncthreads();
  }
  const float ra = red[0] * (1.0f / NM);
  __syncthreads();
  red[tid] = colaux[tid];
  __syncthreads();
  for (int off = 128; off > 0; off >>= 1) {
    if (tid < off) red[tid] += red[tid + off];
    __syncthreads();
  }
  if (tid == 0) out[0] = ra + red[0] * (1.0f / (NB * NE));
}

// ---------------------------------------------------------------------------
extern "C" void kernel_launch(void* const* d_in, const int* in_sizes, int n_in,
                              void* d_out, int out_size, void* d_ws, size_t ws_size,
                              hipStream_t stream) {
  const float* x     = (const float*)d_in[0];
  const float* gamma = (const float*)d_in[1];
  const float* beta  = (const float*)d_in[2];
  const float* W1    = (const float*)d_in[3];
  const float* b1    = (const float*)d_in[4];
  const float* W2    = (const float*)d_in[5];
  const float* b2    = (const float*)d_in[6];
  const float* tau   = (const float*)d_in[7];

  char* ws = (char*)d_ws;
  size_t o = 0;
  _Float16* W1t = (_Float16*)(ws + o); o += (size_t)NH * NH * 2;      // 8 MB
  _Float16* W2t = (_Float16*)(ws + o); o += (size_t)NE * NH * 2;      // 256 KB
  _Float16* xn  = (_Float16*)(ws + o); o += (size_t)NM * NH * 2;      // 32 MB
  _Float16* h   = (_Float16*)(ws + o); o += (size_t)NM * NH * 2;      // 32 MB
  float* assign = (float*)(ws + o);    o += (size_t)NM * NE * 4;      // 2 MB
  float* rowaux = (float*)(ws + o);    o += (size_t)NM * 4;
  float* colaux = (float*)(ws + o);    o += (size_t)NB * NE * 4;

  float* out_idx  = (float*)d_out;            // [B,S,2] indices as float
  float* out_prob = out_idx + (size_t)NM * 2; // [B,S,2]
  float* out_aux  = out_prob + (size_t)NM * 2;

  // 1) Weight transpose + f16 convert
  k_transpose_f32_f16<<<dim3(NH / 64, NH / 64), 256, 0, stream>>>(W1, W1t, NH, NH);
  k_transpose_f32_f16<<<dim3(NE / 64, NH / 64), 256, 0, stream>>>(W2, W2t, NH, NE);

  // 2) LayerNorm
  k_layernorm<<<NM, 256, 0, stream>>>(x, gamma, beta, xn);

  // 3) GEMM1: h = relu(xn @ W1 + b1), f16 out
  k_gemm_wmma<true, true><<<dim3(NH / 64, NM / 128), 256, 0, stream>>>(
      xn, W1t, b1, nullptr, (void*)h, NM, NH, NH);

  // 4) GEMM2: logits = (h @ W2 + b2) / clip(tau), f32 out
  k_gemm_wmma<false, false><<<dim3(NE / 64, NM / 128), 256, 0, stream>>>(
      h, W2t, b2, tau, (void*)assign, NM, NE, NH);

  // 5) scores = exp(logits - rowmax), in place
  k_scores<<<NM / 8, 256, 0, stream>>>(assign);

  // 6) Sinkhorn iterations
  for (int it = 0; it < 10; ++it) {
    k_rownorm<<<NM / 8, 256, 0, stream>>>(assign);
    k_col<false><<<NB * NE, 256, 0, stream>>>(assign, nullptr);
  }

  // 7) Final column stats, then row stats + top-2 outputs
  k_col<true><<<NB * NE, 256, 0, stream>>>(assign, colaux);
  k_final<<<NM / 8, 256, 0, stream>>>(assign, rowaux, out_idx, out_prob);

  // 8) aux loss
  k_aux<<<1, 256, 0, stream>>>(rowaux, colaux, out_aux);

  (void)in_sizes; (void)n_in; (void)out_size; (void)ws_size;
}
